// RPN_41549513622366
// MI455X (gfx1250) — matrix operand order repaired and checked by reading, not compiled, not run on hardware
//
#include <hip/hip_runtime.h>
#include <math.h>

// ---------------------------------------------------------------------------
// Types for CDNA5 WMMA (bf16 held as raw u16 everywhere except the fragments)
// ---------------------------------------------------------------------------
typedef __attribute__((ext_vector_type(16))) __bf16 v16bf;
typedef __attribute__((ext_vector_type(8)))  __bf16 v8bf;
typedef __attribute__((ext_vector_type(8)))  float  v8f;

struct BFx16 { v8bf lo, hi; };   // 32 bytes == v16bf

__device__ __forceinline__ unsigned short f2bf(float f) {
  unsigned u = __builtin_bit_cast(unsigned, f);
  return (unsigned short)((u + 0x7FFFu + ((u >> 16) & 1u)) >> 16);  // RNE
}

#define BATCH 8
#define CIN   256
#define FEAT  256
#define HW    64
#define PADW  66
#define NA    9
#define NANCH (HW * HW * NA)   // 36864
#define PRE   1000
#define POST  300

// ---------------------------------------------------------------------------
// Tensor Data Mover availability (probe-verified builtin; arity differs per
// toolchain — therock headers imply the 6-arg form). Fallback: plain loads.
// ---------------------------------------------------------------------------
#if defined(__has_builtin)
#  if __has_builtin(__builtin_amdgcn_tensor_load_to_lds) && \
      __has_builtin(__builtin_amdgcn_s_wait_tensorcnt)
#    define HAS_TDM 1
#  endif
#endif
#ifndef HAS_TDM
#  define HAS_TDM 0
#endif

#if HAS_TDM
typedef unsigned v4u __attribute__((ext_vector_type(4)));
typedef int      v4i __attribute__((ext_vector_type(4)));
typedef int      v8i __attribute__((ext_vector_type(8)));

// DMA one weight block [tap=9][co=64][ci=32] bf16 (36 KB) into LDS at ldsOff.
// D# per ISA §8.3-8.6: 3-D tile, data_size=2B, dim0(ci): tensor 256/tile 32,
// dim1(co): tensor 256/tile 64/stride 256, dim2(tap): 9 planes/stride 65536.
__device__ __forceinline__ void tdm_load_weights(
    const unsigned short* Wb, unsigned ldsOff, int coBase, int ci0) {
  unsigned long long ga = (unsigned long long)(size_t)Wb +
                          ((unsigned long long)(coBase * 256 + ci0)) * 2ull;
  v4u g0 = { 1u,                                  // count=1, user descriptor
             ldsOff,                               // lds_addr (bytes)
             (unsigned)(ga & 0xFFFFFFFFull),       // global_addr[31:0]
             ((unsigned)((ga >> 32) & 0x1FFFFFFull)) | 0x80000000u }; // type=2
  v8i g1 = { 0x10000,                 // workgroup_mask=0, data_size=1 (2B)
             (int)(256u << 16),       // tensor_dim0 = 256 (low16 @ bits63:48)
             (int)(256u << 16),       // dim0 hi=0 | tensor_dim1 = 256
             (int)(32u << 16),        // dim1 hi=0 | tile_dim0 = 32
             (int)(64u | (9u << 16)), // tile_dim1 = 64, tile_dim2 = 9
             256,                     // tensor_dim0_stride = 256
             0,                       // stride0 hi | stride1 low16 (65536 -> 0)
             1 };                     // tensor_dim1_stride bits47:16 -> 1
  v4i g2 = { 9, 0, 0x10000, 0 };      // tensor_dim2 = 9, dim2_stride = 65536
  v4i g3 = { 0, 0, 0, 0 };
#if __has_include(<hip/amd_detail/amd_gfx1250_TDM.h>)
  v8i g4 = { 0, 0, 0, 0, 0, 0, 0, 0 };
  __builtin_amdgcn_tensor_load_to_lds(g0, g1, g2, g3, g4, 0);
#else
  __builtin_amdgcn_tensor_load_to_lds(g0, g1, g2, g3, 0);
#endif
}
#endif

// Cooperative fallback weight stage (also used if TDM builtin is absent).
__device__ __forceinline__ void fill_sw(unsigned short* sW,
    const unsigned short* Wb, int coBase, int ci0, int tid) {
  for (int e = tid; e < 9 * 64 * 16; e += 256) {
    int k = e >> 10, r = e & 1023, co = r >> 4, cp = r & 15;
    ((unsigned*)sW)[(k * 64 + co) * 16 + cp] =
        *(const unsigned*)(Wb + ((size_t)(k * 256 + coBase + co)) * 256 + ci0 + 2 * cp);
  }
}

// Stage one transposed input slab: sX[x][ci] = Xp[b][ci0+ci][prow][x], x<66.
__device__ __forceinline__ void fill_slab(unsigned short* dst,
    const unsigned short* Xp, int b, int ci0, int prow, int tid) {
  for (int e = tid; e < 33 * 32; e += 256) {
    int ci = e / 33, xp = e - ci * 33;
    unsigned v = *(const unsigned*)(
        Xp + (((size_t)b * CIN + ci0 + ci) * PADW + prow) * PADW + 2 * xp);
    dst[(2 * xp) * 40 + ci]     = (unsigned short)(v & 0xFFFFu);
    dst[(2 * xp + 1) * 40 + ci] = (unsigned short)(v >> 16);
  }
}

// ---------------------------------------------------------------------------
// Prep 1: fp32 input -> zero-padded bf16 image [B][Cin][66][66]
// ---------------------------------------------------------------------------
__global__ __launch_bounds__(256) void pad_input_bf16(
    const float* __restrict__ in, unsigned short* __restrict__ Xp, int total) {
  int i = blockIdx.x * 256 + threadIdx.x;
  if (i >= total) return;
  int px = i % PADW; int t = i / PADW;
  int py = t % PADW; t /= PADW;
  int c  = t % CIN;  int b = t / CIN;
  float v = 0.f;
  if (py >= 1 && py <= HW && px >= 1 && px <= HW)
    v = in[(((size_t)b * CIN + c) * HW + (py - 1)) * HW + (px - 1)];
  Xp[i] = f2bf(v);
}

// ---------------------------------------------------------------------------
// Prep 2: conv weights (Cout,Cin,3,3) fp32 -> bf16 [tap][Cout][Cin]
// ---------------------------------------------------------------------------
__global__ __launch_bounds__(256) void conv_w_bf16(
    const float* __restrict__ w, unsigned short* __restrict__ Wb, int total) {
  int i = blockIdx.x * 256 + threadIdx.x;
  if (i >= total) return;
  int ci = i % CIN; int t = i / CIN;
  int co = t % FEAT; int k = t / FEAT;      // k = kh*3 + kw
  int kh = k / 3, kw = k % 3;
  Wb[i] = f2bf(w[(((size_t)co * CIN + ci) * 3 + kh) * 3 + kw]);
}

// ---------------------------------------------------------------------------
// Main 3x3 conv + bias + ReLU as implicit GEMM with v_wmma_f32_16x16x32_bf16.
// Double-buffered input slab (one barrier/step), TDM-staged weights.
// Grid: (h=64, coBlk=4, b=8), 256 threads = 8 waves.
// ---------------------------------------------------------------------------
__global__ __launch_bounds__(256) void conv3_wmma(
    const unsigned short* __restrict__ Xp,   // [B][256][66][66] bf16
    const unsigned short* __restrict__ Wb,   // [9][256][256]    bf16
    const float* __restrict__ bias,
    float* __restrict__ Y)                   // [B][256][64][64] f32
{
  __shared__ __align__(16) unsigned short sW[9 * 64 * 32];   // 36 KB
  __shared__ __align__(16) unsigned short sX[2][66 * 40];    // 2 x 5.28 KB

  const int tid     = threadIdx.x;
  const int lane    = tid & 31;
  const int wv      = tid >> 5;
  const int coSub   = wv & 3;
  const int pixHalf = wv >> 2;
  const int h       = blockIdx.x;
  const int coBase  = blockIdx.y * 64;
  const int b       = blockIdx.z;
  const int m       = lane & 15;
  const int laneHiA = (lane < 16) ? 0 : 16;  // A-matrix K half (contiguous k = 16*(lane/16) + s)
  const int laneHiB = (lane < 16) ? 0 : 16;  // B-matrix K sub-block

  v8f acc0 = {}; v8f acc1 = {};

  // ---- prologue: weights(cc=0) + slab(step 0)
#if HAS_TDM
  if (tid < 32)
    tdm_load_weights(Wb, (unsigned)(size_t)(void*)sW, coBase, 0);
#endif
  fill_slab(&sX[0][0], Xp, b, 0, h, tid);
#if HAS_TDM
  if (tid < 32) __builtin_amdgcn_s_wait_tensorcnt(0);
#else
  fill_sw(sW, Wb, coBase, 0, tid);
#endif
  __syncthreads();

  for (int s = 0; s < 24; ++s) {             // s = cc*3 + dh
    const int dh  = s % 3;
    const int cur = s & 1;
    if (s + 1 < 24) {                        // prefetch next slab (other buffer)
      const int s1 = s + 1;
      fill_slab(&sX[1 - cur][0], Xp, b, (s1 / 3) * 32, h + (s1 % 3), tid);
    }
    const unsigned short* xb = &sX[cur][0];
    #pragma unroll
    for (int dw = 0; dw < 3; ++dw) {
      const int k = dh * 3 + dw;
      const v8bf* ap = (const v8bf*)(sW +
          (size_t)((k * 64 + coSub * 16 + m) * 32 + laneHiA));
      BFx16 at; at.lo = ap[0]; at.hi = ap[1];       // K {0..15}(+hi)
      v16bf av = __builtin_bit_cast(v16bf, at);

      const v8bf* bp0 = (const v8bf*)(xb +
          (size_t)((pixHalf * 32 + m + dw) * 40 + laneHiB));
      BFx16 bt0; bt0.lo = bp0[0]; bt0.hi = bp0[1];
      acc0 = __builtin_amdgcn_wmma_f32_16x16x32_bf16(
          false, av, false, __builtin_bit_cast(v16bf, bt0),
          (short)0, acc0, false, false);

      const v8bf* bp1 = (const v8bf*)(xb +
          (size_t)((pixHalf * 32 + 16 + m + dw) * 40 + laneHiB));
      BFx16 bt1; bt1.lo = bp1[0]; bt1.hi = bp1[1];
      acc1 = __builtin_amdgcn_wmma_f32_16x16x32_bf16(
          false, av, false, __builtin_bit_cast(v16bf, bt1),
          (short)0, acc1, false, false);
    }
    if (dh == 2 && s + 1 < 24) {             // Cin-chunk boundary: reload sW
      __syncthreads();                       // all waves done reading sW(cc)
#if HAS_TDM
      if (tid < 32) {
        tdm_load_weights(Wb, (unsigned)(size_t)(void*)sW, coBase,
                         ((s + 1) / 3) * 32);
        __builtin_amdgcn_s_wait_tensorcnt(0);
      }
#else
      fill_sw(sW, Wb, coBase, ((s + 1) / 3) * 32, tid);
#endif
      __syncthreads();                       // new sW + prefetched slab visible
    } else {
      __syncthreads();                       // swap double buffers
    }
  }

  // Epilogue: C/D layout — VGPR r: lanes0-15 M=r, lanes16-31 M=r+8; N=lane&15
  const int x0 = pixHalf * 32 + m;
  #pragma unroll
  for (int r = 0; r < 8; ++r) {
    int co = coBase + coSub * 16 + r + ((lane < 16) ? 0 : 8);
    float bv = bias[co];
    size_t o = (((size_t)b * FEAT + co) * HW + h) * HW;
    Y[o + x0]      = fmaxf(acc0[r] + bv, 0.f);
    Y[o + x0 + 16] = fmaxf(acc1[r] + bv, 0.f);
  }
}

// ---------------------------------------------------------------------------
// 1x1 heads (clf 18ch + reg 36ch) + softmax + anchor decode.
// One thread per (b, pixel, anchor). Head weights staged in LDS (55 KB).
// ---------------------------------------------------------------------------
__global__ __launch_bounds__(256) void heads_decode(
    const float* __restrict__ Y, const float* __restrict__ wclf,
    const float* __restrict__ bclf, const float* __restrict__ wreg,
    const float* __restrict__ breg, const float* __restrict__ anchors,
    float* __restrict__ scores, float* __restrict__ props)
{
  __shared__ float sW[54 * 256];
  __shared__ float sB[54];
  int tid = threadIdx.x;
  for (int e = tid; e < 18 * 256; e += 256) sW[e] = wclf[e];
  for (int e = tid; e < 36 * 256; e += 256) sW[18 * 256 + e] = wreg[e];
  if (tid < 18) sB[tid] = bclf[tid];
  if (tid < 36) sB[18 + tid] = breg[tid];
  __syncthreads();

  int g = blockIdx.x * 256 + tid;            // 0 .. B*NANCH-1 (exact grid)
  int b = g / NANCH; int r = g - b * NANCH;
  int p = r / NA;    int a = r - p * NA;
  int h = p >> 6;    int w = p & 63;

  const float* fb = Y + (((size_t)b * FEAT) * HW + h) * HW + w; // ch stride 4096
  const float* w0 = sW + a * 256;
  const float* w1 = sW + (9 + a) * 256;
  const float* wr = sW + (18 + a * 4) * 256;
  float s0 = sB[a], s1 = sB[9 + a];
  float d0 = sB[18 + a * 4], d1 = sB[18 + a * 4 + 1];
  float d2 = sB[18 + a * 4 + 2], d3 = sB[18 + a * 4 + 3];
  #pragma unroll 4
  for (int c = 0; c < 256; ++c) {
    float f = fb[(size_t)c * (HW * HW)];
    s0 = fmaf(f, w0[c], s0);          s1 = fmaf(f, w1[c], s1);
    d0 = fmaf(f, wr[c], d0);          d1 = fmaf(f, wr[c + 256], d1);
    d2 = fmaf(f, wr[c + 512], d2);    d3 = fmaf(f, wr[c + 768], d3);
  }
  float score = 1.f / (1.f + __expf(s0 - s1));   // softmax class-1 prob

  const float* an = anchors + (size_t)r * 4;
  float aw = an[2] - an[0], ah = an[3] - an[1];
  float cx = 0.5f * (an[0] + an[2]), cy = 0.5f * (an[1] + an[3]);
  float pcx = d0 * aw + cx, pcy = d1 * ah + cy;
  float pw = __expf(d2) * aw, ph = __expf(d3) * ah;
  float x1 = fminf(fmaxf(pcx - 0.5f * pw, 0.f), 1023.f);
  float y1 = fminf(fmaxf(pcy - 0.5f * ph, 0.f), 1023.f);
  float x2 = fminf(fmaxf(pcx + 0.5f * pw, 0.f), 1023.f);
  float y2 = fminf(fmaxf(pcy + 0.5f * ph, 0.f), 1023.f);

  scores[g] = score;
  float* pp = props + (size_t)g * 4;
  pp[0] = x1; pp[1] = y1; pp[2] = x2; pp[3] = y2;
}

// ---------------------------------------------------------------------------
// Per-batch top-1000: two-level 11-bit radix select + 1024-wide bitonic sort.
// ---------------------------------------------------------------------------
__device__ __forceinline__ unsigned sortkey(float s) {
  unsigned u = __builtin_bit_cast(unsigned, s);
  return (u & 0x80000000u) ? ~u : (u | 0x80000000u);
}

__global__ __launch_bounds__(1024) void topk_sort(
    const float* __restrict__ scores, const float* __restrict__ props,
    float* __restrict__ topBoxes)
{
  __shared__ unsigned hist[2048];
  __shared__ unsigned sKey[1024];
  __shared__ unsigned sIdx[1024];
  __shared__ unsigned sBB, sCA, sBB2, sRem2, sCnt, sTie;
  int tid = threadIdx.x;
  int b   = blockIdx.x;
  const float* sc = scores + (size_t)b * NANCH;

  for (int e = tid; e < 2048; e += 1024) hist[e] = 0;
  if (tid == 0) { sCnt = 0; sTie = 0; }
  __syncthreads();
  for (int n = tid; n < NANCH; n += 1024)
    atomicAdd(&hist[sortkey(sc[n]) >> 21], 1u);
  __syncthreads();
  if (tid == 0) {
    unsigned cum = 0; int bb = 0;
    for (int i = 2047; i >= 0; --i) {
      if (cum + hist[i] >= PRE) { bb = i; break; }
      cum += hist[i];
    }
    sBB = (unsigned)bb; sCA = cum;
  }
  __syncthreads();
  unsigned bb = sBB, cA = sCA;

  for (int e = tid; e < 2048; e += 1024) hist[e] = 0;
  __syncthreads();
  for (int n = tid; n < NANCH; n += 1024) {
    unsigned k = sortkey(sc[n]);
    if ((k >> 21) == bb) atomicAdd(&hist[(k >> 10) & 0x7FFu], 1u);
  }
  __syncthreads();
  if (tid == 0) {
    unsigned rem = PRE - cA, cum = 0; int bb2 = 0;
    for (int i = 2047; i >= 0; --i) {
      if (cum + hist[i] >= rem) { bb2 = i; break; }
      cum += hist[i];
    }
    sBB2 = (unsigned)bb2; sRem2 = rem - cum;
  }
  __syncthreads();
  unsigned bb2 = sBB2, rem2 = sRem2;

  for (int n = tid; n < NANCH; n += 1024) {
    unsigned k = sortkey(sc[n]);
    unsigned kb = k >> 21;
    bool take = false;
    if (kb > bb) take = true;
    else if (kb == bb) {
      unsigned sb = (k >> 10) & 0x7FFu;
      if (sb > bb2) take = true;
      else if (sb == bb2) take = (atomicAdd(&sTie, 1u) < rem2);
    }
    if (take) {
      unsigned p = atomicAdd(&sCnt, 1u);
      if (p < 1024) { sKey[p] = k; sIdx[p] = (unsigned)n; }
    }
  }
  __syncthreads();
  unsigned cnt = sCnt > 1024u ? 1024u : sCnt;
  if ((unsigned)tid >= cnt) { sKey[tid] = 0; sIdx[tid] = 0; }
  __syncthreads();

  // bitonic sort, descending by key
  for (unsigned size = 2; size <= 1024; size <<= 1) {
    for (unsigned stride = size >> 1; stride > 0; stride >>= 1) {
      unsigned partner = (unsigned)tid ^ stride;
      if (partner > (unsigned)tid) {
        bool desc = (((unsigned)tid & size) == 0);
        unsigned ka = sKey[tid], kb2 = sKey[partner];
        bool sw = desc ? (ka < kb2) : (ka > kb2);
        if (sw) {
          sKey[tid] = kb2; sKey[partner] = ka;
          unsigned ia = sIdx[tid]; sIdx[tid] = sIdx[partner]; sIdx[partner] = ia;
        }
      }
      __syncthreads();
    }
  }

  float* ob = topBoxes + ((size_t)b * 1024 + tid) * 4;
  if (tid < PRE) {
    const float* pp = props + ((size_t)b * NANCH + sIdx[tid]) * 4;
    ob[0] = pp[0]; ob[1] = pp[1]; ob[2] = pp[2]; ob[3] = pp[3];
  } else {
    ob[0] = 0.f; ob[1] = 0.f; ob[2] = 0.f; ob[3] = 0.f;
  }
}

// ---------------------------------------------------------------------------
// Per-batch greedy NMS (IoU > 0.7) over the sorted top-1000, then compact
// first 300 survivors into rois (col0 = batch idx for ALL rows).
// ---------------------------------------------------------------------------
__global__ __launch_bounds__(256) void nms_compact(
    const float* __restrict__ topBoxes, float* __restrict__ out)
{
  __shared__ float bx[PRE * 4];
  __shared__ int   keep[PRE];
  __shared__ int   pos[PRE];
  int tid = threadIdx.x;
  int b   = blockIdx.x;
  for (int e = tid; e < PRE * 4; e += 256)
    bx[e] = topBoxes[((size_t)b * 1024) * 4 + e];
  for (int e = tid; e < PRE; e += 256) keep[e] = 1;
  __syncthreads();

  for (int i = 0; i < PRE - 1; ++i) {
    if (keep[i]) {
      float ax1 = bx[i * 4], ay1 = bx[i * 4 + 1];
      float ax2 = bx[i * 4 + 2], ay2 = bx[i * 4 + 3];
      float aa = (ax2 - ax1) * (ay2 - ay1);
      for (int j = i + 1 + tid; j < PRE; j += 256) {
        if (keep[j]) {
          float x1 = fmaxf(ax1, bx[j * 4]),     y1 = fmaxf(ay1, bx[j * 4 + 1]);
          float x2 = fminf(ax2, bx[j * 4 + 2]), y2 = fminf(ay2, bx[j * 4 + 3]);
          float inter = fmaxf(x2 - x1, 0.f) * fmaxf(y2 - y1, 0.f);
          float ab = (bx[j * 4 + 2] - bx[j * 4]) * (bx[j * 4 + 3] - bx[j * 4 + 1]);
          if (inter / (aa + ab - inter) > 0.7f) keep[j] = 0;
        }
      }
    }
    __syncthreads();
  }

  if (tid == 0) {
    int c = 0;
    for (int j = 0; j < PRE; ++j) { pos[j] = keep[j] ? c : -1; c += keep[j]; }
  }
  for (int e = tid; e < POST * 5; e += 256)
    out[(size_t)b * POST * 5 + e] = ((e % 5) == 0) ? (float)b : 0.f;
  __syncthreads();
  for (int j = tid; j < PRE; j += 256) {
    int p = pos[j];
    if (p >= 0 && p < POST) {
      float* o = out + ((size_t)b * POST + p) * 5;
      o[1] = bx[j * 4];     o[2] = bx[j * 4 + 1];
      o[3] = bx[j * 4 + 2]; o[4] = bx[j * 4 + 3];
    }
  }
}

// ---------------------------------------------------------------------------
extern "C" void kernel_launch(void* const* d_in, const int* in_sizes, int n_in,
                              void* d_out, int out_size, void* d_ws, size_t ws_size,
                              hipStream_t stream) {
  (void)in_sizes; (void)n_in; (void)out_size; (void)ws_size;
  const float* inp     = (const float*)d_in[0];
  const float* anchors = (const float*)d_in[2];
  const float* wconv   = (const float*)d_in[3];
  const float* bconv   = (const float*)d_in[4];
  const float* wclf    = (const float*)d_in[5];
  const float* bclf    = (const float*)d_in[6];
  const float* wreg    = (const float*)d_in[7];
  const float* breg    = (const float*)d_in[8];
  float* out = (float*)d_out;

  char* ws = (char*)d_ws;
  size_t off = 0;
  auto carve = [&](size_t bytes) {
    size_t o = off;
    off = (off + bytes + 255) & ~(size_t)255;
    return o;
  };
  unsigned short* Xp = (unsigned short*)(ws + carve((size_t)BATCH * CIN * PADW * PADW * 2));
  unsigned short* Wb = (unsigned short*)(ws + carve((size_t)9 * FEAT * CIN * 2));
  float* Yf        = (float*)(ws + carve((size_t)BATCH * FEAT * HW * HW * 4));
  float* scoresP   = (float*)(ws + carve((size_t)BATCH * NANCH * 4));
  float* propsP    = (float*)(ws + carve((size_t)BATCH * NANCH * 4 * 4));
  float* topBoxesP = (float*)(ws + carve((size_t)BATCH * 1024 * 4 * 4));

  int totX = BATCH * CIN * PADW * PADW;            // 8,921,088 (multiple of 256)
  pad_input_bf16<<<totX / 256, 256, 0, stream>>>(inp, Xp, totX);
  int totW = 9 * FEAT * CIN;                       // 589,824
  conv_w_bf16<<<totW / 256, 256, 0, stream>>>(wconv, Wb, totW);

  dim3 g1(HW, FEAT / 64, BATCH);                   // (64,4,8)
  conv3_wmma<<<g1, 256, 0, stream>>>(Xp, Wb, bconv, Yf);

  heads_decode<<<(BATCH * NANCH) / 256, 256, 0, stream>>>(
      Yf, wclf, bclf, wreg, breg, anchors, scoresP, propsP);

  topk_sort<<<BATCH, 1024, 0, stream>>>(scoresP, propsP, topBoxesP);
  nms_compact<<<BATCH, 256, 0, stream>>>(topBoxesP, out);
}
